// SelfAttention_65996467470776
// MI455X (gfx1250) — matrix operand order repaired and checked by reading, not compiled, not run on hardware
//
#include <hip/hip_runtime.h>
#include <math.h>

// Flash attention, B x L x D = 8 x 4096 x 64, fp32 in/out, causal + zero-key mask.
// WMMA f16 16x16x32: 8x QK^T + 8x PV + 2x rowsum(P x ones) per 64-key tile.
// Online softmax, double-buffered LDS tiles, packed b32 fragment loads,
// cvt_pkrtz conversions, branchless masking.

typedef __attribute__((ext_vector_type(16))) _Float16     v16h;
typedef __attribute__((ext_vector_type(2)))  __fp16       f16x2;
typedef __attribute__((ext_vector_type(8)))  float        v8f;
typedef __attribute__((ext_vector_type(8)))  unsigned int v8u;

#define TILE_Q 64      // query rows per block (16 per wave)
#define TILE_K 64      // keys per iteration (2x WMMA K)
#define HEAD_D 64
#define NT     128     // 4 waves (wave32)
#define KROW   68      // halfs per K-tile row: 136B, 34 dwords
#define VROW   68      // halfs per transposed-V row
#define PROW   66      // halfs per P-staging row: 132B, 33 dwords

__device__ __forceinline__ float red_max16(float v) {
    v = fmaxf(v, __shfl_xor(v, 1, 32));
    v = fmaxf(v, __shfl_xor(v, 2, 32));
    v = fmaxf(v, __shfl_xor(v, 4, 32));
    v = fmaxf(v, __shfl_xor(v, 8, 32));
    return v;
}
// A-matrix (16x32 f16) K-pair base for vgpr-pair i, lane-half h (ISA 7.12.2)
__device__ __forceinline__ int a_kbase(int i, int h) {
    return (i < 4 ? 2 * i : 16 + 2 * (i - 4)) + (h ? 8 : 0);
}
// B-matrix (32x16 f16) K-pair base for vgpr-pair i, lane-half h
__device__ __forceinline__ int b_kbase(int i, int h) {
    return (i < 4 ? 2 * i : 16 + 2 * (i - 4)) + (h ? 8 : 0);
}
__device__ __forceinline__ unsigned int pkh(float a, float b) {
    f16x2 p = __builtin_amdgcn_cvt_pkrtz(a, b);
    return __builtin_bit_cast(unsigned int, p);
}

__global__ __launch_bounds__(NT)
void fa_wmma_kernel(const float* __restrict__ Qg,
                    const float* __restrict__ Kg,
                    const float* __restrict__ Vg,
                    const int*   __restrict__ amask,
                    float*       __restrict__ Og,
                    int L) {
    __shared__ _Float16 Kl[2][TILE_K][KROW];   // K tile, row-major [key][d]
    __shared__ _Float16 Vt[2][HEAD_D][VROW];   // V tile, transposed [d][key]
    __shared__ _Float16 Pl[4][16][PROW];       // per-wave P staging (16 x 64)
    __shared__ float    kflag[2][TILE_K];      // 0.0 iff key row sums to exactly 0

    const int tid  = threadIdx.x;
    const int lane = tid & 31;
    const int w    = tid >> 5;
    const int hsel = lane >> 4;                // lane half
    const int n    = lane & 15;                // B/C/D column; A row

    const int b      = blockIdx.y;
    const int qblk   = blockIdx.x;
    const int qbase  = qblk * TILE_Q + w * 16;
    const int causal = amask[0];
    const float NEG   = -4294967295.0f;        // -(2^32) + 1
    const float scale = 0.125f;                // 1/sqrt(64)
    const size_t bLD  = (size_t)b * L * HEAD_D;
    const int qrbase  = qbase + hsel * 8;      // C/D row base for this lane

    // tile loader mapping: each thread owns a contiguous 32-float half-row of one key
    const int lkey   = tid >> 1;               // key 0..63
    const int lchunk = (tid & 1) * 32;         // d offset 0/32

    // ---- Q fragments (A layout), scale folded in ----
    v16h aQ0, aQ1;
    {
        const float* qrow = Qg + bLD + (size_t)(qbase + n) * HEAD_D;
        v8u u0, u1;
#pragma unroll
        for (int i = 0; i < 8; ++i) {
            int ka = a_kbase(i, hsel);
            u0[i] = pkh(qrow[ka] * scale, qrow[ka + 1] * scale);
            u1[i] = pkh(qrow[32 + ka] * scale, qrow[32 + ka + 1] * scale);
        }
        aQ0 = __builtin_bit_cast(v16h, u0);
        aQ1 = __builtin_bit_cast(v16h, u1);
    }
    v16h bOnes;
    {
        v8u u;
#pragma unroll
        for (int i = 0; i < 8; ++i) u[i] = 0x3C003C00u;   // (1.0h, 1.0h)
        bOnes = __builtin_bit_cast(v16h, u);
    }

    float rowm[8];
    v8f oacc[4], osum;
    v8f zero = {};
#pragma unroll
    for (int j = 0; j < 8; ++j) rowm[j] = -3.0e38f;
#pragma unroll
    for (int c = 0; c < 4; ++c) oacc[c] = zero;
    osum = zero;

    const int numKB = causal ? (qblk + 1) : (L / TILE_K);

    // ping-pong buffer pointers
    _Float16 (*Kcur)[KROW] = Kl[0], (*Knxt)[KROW] = Kl[1];
    _Float16 (*Vcur)[VROW] = Vt[0], (*Vnxt)[VROW] = Vt[1];
    float *kfcur = kflag[0], *kfnxt = kflag[1];

    // ---- prologue: stage tile 0 into buffer 0 ----
    float4 kq[8], vq[8];
    {
        const float4* kp = (const float4*)(Kg + bLD + (size_t)lkey * HEAD_D + lchunk);
        const float4* vp = (const float4*)(Vg + bLD + (size_t)lkey * HEAD_D + lchunk);
#pragma unroll
        for (int c = 0; c < 8; ++c) { kq[c] = kp[c]; vq[c] = vp[c]; }
    }
    {
        float s = 0.0f;
#pragma unroll
        for (int c = 0; c < 8; ++c) {
            *(unsigned int*)&Kcur[lkey][lchunk + 4 * c]     = pkh(kq[c].x, kq[c].y);
            *(unsigned int*)&Kcur[lkey][lchunk + 4 * c + 2] = pkh(kq[c].z, kq[c].w);
            Vcur[lchunk + 4 * c + 0][lkey] = (_Float16)vq[c].x;
            Vcur[lchunk + 4 * c + 1][lkey] = (_Float16)vq[c].y;
            Vcur[lchunk + 4 * c + 2][lkey] = (_Float16)vq[c].z;
            Vcur[lchunk + 4 * c + 3][lkey] = (_Float16)vq[c].w;
            s += kq[c].x + kq[c].y + kq[c].z + kq[c].w;
        }
        s += __shfl_xor(s, 1, 32);
        if ((tid & 1) == 0) kfcur[lkey] = (s == 0.0f) ? 0.0f : 1.0f;
    }

    // key index substitute when causal masking is off: always valid
    const int ekbase = causal ? 0 : (int)0x80000000;

    for (int kbi = 0; kbi < numKB; ++kbi) {
        const int kb = kbi * TILE_K;
        const bool havenext = (kbi + 1 < numKB);

        __syncthreads();   // tile kbi visible in current buffers

        // ---- issue next tile's global loads early (latency hiding) ----
        if (havenext) {
            const size_t g = bLD + (size_t)(kb + TILE_K + lkey) * HEAD_D + lchunk;
            const float4* kp = (const float4*)(Kg + g);
            const float4* vp = (const float4*)(Vg + g);
#pragma unroll
            for (int c = 0; c < 8; ++c) { kq[c] = kp[c]; vq[c] = vp[c]; }
        }

        {
            // ---- S = Q K^T : four 16x16 key sub-tiles, K-dim 64 = 2 WMMA chunks ----
            v8f s[4];
#pragma unroll
            for (int nb = 0; nb < 4; ++nb) {
                v8u u;
#pragma unroll
                for (int i = 0; i < 8; ++i)
                    u[i] = *(const unsigned int*)&Kcur[nb * 16 + n][b_kbase(i, hsel)];
                v16h bk = __builtin_bit_cast(v16h, u);
                s[nb] = __builtin_amdgcn_wmma_f32_16x16x32_f16(false, aQ0, false, bk,
                                                               (short)0, zero, false, false);
#pragma unroll
                for (int i = 0; i < 8; ++i)
                    u[i] = *(const unsigned int*)&Kcur[nb * 16 + n][b_kbase(i, hsel) + 32];
                bk = __builtin_bit_cast(v16h, u);
                s[nb] = __builtin_amdgcn_wmma_f32_16x16x32_f16(false, aQ1, false, bk,
                                                               (short)0, s[nb], false, false);
            }

            // ---- branchless masking + online softmax ----
            // effective key index: INT_MAX for padded keys (never <= qr),
            // INT_MIN when causal masking disabled (always <= qr), else real index.
            int ek[4];
#pragma unroll
            for (int nb = 0; nb < 4; ++nb) {
                float kf = kfcur[nb * 16 + n];
                ek[nb] = (kf == 0.0f) ? 0x7FFFFFFF : (ekbase + kb + nb * 16 + n);
            }

#pragma unroll
            for (int j = 0; j < 8; ++j) {
                const int qr = qrbase + j;
                float v0 = (ek[0] <= qr) ? s[0][j] : NEG;
                float v1 = (ek[1] <= qr) ? s[1][j] : NEG;
                float v2 = (ek[2] <= qr) ? s[2][j] : NEG;
                float v3 = (ek[3] <= qr) ? s[3][j] : NEG;

                float lm   = fmaxf(fmaxf(v0, v1), fmaxf(v2, v3));
                float mnew = fmaxf(rowm[j], red_max16(lm));
                float fac  = __expf(rowm[j] - mnew);
                rowm[j] = mnew;
                float p0 = __expf(v0 - mnew);
                float p1 = __expf(v1 - mnew);
                float p2 = __expf(v2 - mnew);
                float p3 = __expf(v3 - mnew);
                osum[j]    *= fac;
                oacc[0][j] *= fac; oacc[1][j] *= fac;
                oacc[2][j] *= fac; oacc[3][j] *= fac;

                const int r = j + hsel * 8;
                Pl[w][r][n]      = (_Float16)p0;
                Pl[w][r][16 + n] = (_Float16)p1;
                Pl[w][r][32 + n] = (_Float16)p2;
                Pl[w][r][48 + n] = (_Float16)p3;
            }

            asm volatile("s_wait_dscnt 0" ::: "memory");  // same-wave cross-lane LDS RAW

            // ---- reload P in A layout (packed), two K-chunks ----
            v16h aP0, aP1;
            {
                v8u u0, u1;
#pragma unroll
                for (int i = 0; i < 8; ++i) {
                    int ka = a_kbase(i, hsel);
                    u0[i] = *(const unsigned int*)&Pl[w][n][ka];
                    u1[i] = *(const unsigned int*)&Pl[w][n][ka + 32];
                }
                aP0 = __builtin_bit_cast(v16h, u0);
                aP1 = __builtin_bit_cast(v16h, u1);
            }

            // ---- O += P V (4 d-tiles x 2 K-chunks) and rowsum(P) via P x ones ----
#pragma unroll
            for (int c = 0; c < 4; ++c) {
                v8u u;
#pragma unroll
                for (int i = 0; i < 8; ++i)
                    u[i] = *(const unsigned int*)&Vcur[c * 16 + n][b_kbase(i, hsel)];
                v16h bv = __builtin_bit_cast(v16h, u);
                oacc[c] = __builtin_amdgcn_wmma_f32_16x16x32_f16(false, aP0, false, bv,
                                                                 (short)0, oacc[c], false, false);
#pragma unroll
                for (int i = 0; i < 8; ++i)
                    u[i] = *(const unsigned int*)&Vcur[c * 16 + n][b_kbase(i, hsel) + 32];
                bv = __builtin_bit_cast(v16h, u);
                oacc[c] = __builtin_amdgcn_wmma_f32_16x16x32_f16(false, aP1, false, bv,
                                                                 (short)0, oacc[c], false, false);
            }
            osum = __builtin_amdgcn_wmma_f32_16x16x32_f16(false, aP0, false, bOnes,
                                                          (short)0, osum, false, false);
            osum = __builtin_amdgcn_wmma_f32_16x16x32_f16(false, aP1, false, bOnes,
                                                          (short)0, osum, false, false);
        }

        // ---- convert + store next tile into the other buffer ----
        if (havenext) {
            float s = 0.0f;
#pragma unroll
            for (int c = 0; c < 8; ++c) {
                *(unsigned int*)&Knxt[lkey][lchunk + 4 * c]     = pkh(kq[c].x, kq[c].y);
                *(unsigned int*)&Knxt[lkey][lchunk + 4 * c + 2] = pkh(kq[c].z, kq[c].w);
                Vnxt[lchunk + 4 * c + 0][lkey] = (_Float16)vq[c].x;
                Vnxt[lchunk + 4 * c + 1][lkey] = (_Float16)vq[c].y;
                Vnxt[lchunk + 4 * c + 2][lkey] = (_Float16)vq[c].z;
                Vnxt[lchunk + 4 * c + 3][lkey] = (_Float16)vq[c].w;
                s += kq[c].x + kq[c].y + kq[c].z + kq[c].w;
            }
            s += __shfl_xor(s, 1, 32);
            if ((tid & 1) == 0) kfnxt[lkey] = (s == 0.0f) ? 0.0f : 1.0f;
        }

        // swap ping-pong pointers
        { auto t = Kcur; Kcur = Knxt; Knxt = t; }
        { auto t = Vcur; Vcur = Vnxt; Vnxt = t; }
        { auto t = kfcur; kfcur = kfnxt; kfnxt = t; }
    }

    // ---- normalize and store ----
#pragma unroll
    for (int c = 0; c < 4; ++c) {
#pragma unroll
        for (int j = 0; j < 8; ++j) {
            int qr = qrbase + j;
            Og[bLD + (size_t)qr * HEAD_D + c * 16 + n] = oacc[c][j] / osum[j];
        }
    }
}

extern "C" void kernel_launch(void* const* d_in, const int* in_sizes, int n_in,
                              void* d_out, int out_size, void* d_ws, size_t ws_size,
                              hipStream_t stream) {
    const float* Q = (const float*)d_in[0];
    const float* K = (const float*)d_in[1];
    const float* V = (const float*)d_in[2];
    const int*   M = (const int*)d_in[3];
    float*       O = (float*)d_out;

    const int L = 4096, D = 64;
    const int B = in_sizes[0] / (L * D);

    dim3 grid(L / TILE_Q, B);
    dim3 block(NT);
    fa_wmma_kernel<<<grid, block, 0, stream>>>(Q, K, V, M, O, L);
}
